// CoGNN_55748675502551
// MI455X (gfx1250) — matrix operand order repaired and checked by reading, not compiled, not run on hardware
//
#include <hip/hip_runtime.h>
#include <hip/hip_bf16.h>
#include <math.h>

typedef __attribute__((ext_vector_type(16))) _Float16 v16h;
typedef __attribute__((ext_vector_type(4)))  _Float16 v4h;
typedef __attribute__((ext_vector_type(8)))  float    v8f;

// ---------------------------------------------------------------- utilities
__global__ void fill_f32(float* __restrict__ p, float v, int n) {
    int i = blockIdx.x * blockDim.x + threadIdx.x;
    if (i < n) p[i] = v;
}
__global__ void fill_u32(unsigned* __restrict__ p, unsigned v, int n) {
    int i = blockIdx.x * blockDim.x + threadIdx.x;
    if (i < n) p[i] = v;
}
__global__ void fill_f16(_Float16* __restrict__ p, int n) {
    int i = blockIdx.x * blockDim.x + threadIdx.x;
    if (i < n) p[i] = (_Float16)0.0f;
}
__global__ void cvt_f16(const float* __restrict__ in, _Float16* __restrict__ out, int n) {
    int i = blockIdx.x * blockDim.x + threadIdx.x;
    if (i < n) out[i] = (_Float16)in[i];
}
// pad-convert rows: src [rows, ksrc] f32 -> dst [rows, kdst] f16, zero-pad k in [ksrc, kdst)
__global__ void padcvt_f16(const float* __restrict__ src, _Float16* __restrict__ dst,
                           int rows, int ksrc, int kdst) {
    int i = blockIdx.x * blockDim.x + threadIdx.x;
    if (i >= rows * kdst) return;
    int r = i / kdst, k = i - r * kdst;
    dst[i] = (k < ksrc) ? (_Float16)src[(size_t)r * ksrc + k] : (_Float16)0.0f;
}

__device__ __forceinline__ unsigned long long splitmix64(unsigned long long x) {
    x += 0x9E3779B97F4A7C15ULL;
    x = (x ^ (x >> 30)) * 0xBF58476D1CE4E5B9ULL;
    x = (x ^ (x >> 27)) * 0x94D049BB133111EBULL;
    return x ^ (x >> 31);
}
__device__ __forceinline__ float gumbelf(unsigned long long h) {
    float u = (float)(h >> 40) * (1.0f / 16777216.0f);   // 24-bit uniform [0,1)
    return -__logf(-__logf(u + 1e-12f) + 1e-12f);
}
__device__ __forceinline__ unsigned enc_f(float f) {
    unsigned b = __float_as_uint(f);
    return b ^ ((b & 0x80000000u) ? 0xFFFFFFFFu : 0x80000000u);
}
__device__ __forceinline__ float dec_f(unsigned k) {
    unsigned b = k ^ ((k & 0x80000000u) ? 0x80000000u : 0xFFFFFFFFu);
    return __uint_as_float(b);
}

// ---------------------------------------------------------------- LayerNorm (one wave32 per node, 64 ch)
__global__ void ln_kernel(const float* __restrict__ h, const float* __restrict__ g,
                          const float* __restrict__ b, float* __restrict__ out, int n) {
    int node = blockIdx.x * 8 + (threadIdx.x >> 5);
    int lane = threadIdx.x & 31;
    if (node >= n) return;                       // whole wave exits together
    float x0 = h[node * 64 + lane];
    float x1 = h[node * 64 + lane + 32];
    float s = x0 + x1;
#pragma unroll
    for (int m = 16; m >= 1; m >>= 1) s += __shfl_xor(s, m, 32);
    float mean = s * (1.0f / 64.0f);
    float d0 = x0 - mean, d1 = x1 - mean;
    float vs = d0 * d0 + d1 * d1;
#pragma unroll
    for (int m = 16; m >= 1; m >>= 1) vs += __shfl_xor(vs, m, 32);
    float inv = rsqrtf(vs * (1.0f / 64.0f) + 1e-5f);
    out[node * 64 + lane]      = d0 * inv * g[lane]      + b[lane];
    out[node * 64 + lane + 32] = d1 * inv * g[lane + 32] + b[lane + 32];
}

// ---------------------------------------------------------------- edge aggregation (fused edge-encoder), 64 ch
__global__ void agg64_kernel(const float* __restrict__ hsrc, const float* __restrict__ ea,
                             const float* __restrict__ We, const float* __restrict__ be,
                             const int* __restrict__ u, const int* __restrict__ v,
                             const float* __restrict__ keep_in, const float* __restrict__ keep_out,
                             float* __restrict__ agg, int E) {
    int gid = blockIdx.x * blockDim.x + threadIdx.x;
    if (gid >= E * 64) return;
    int e = gid >> 6, c = gid & 63;
    float env = be[c];
#pragma unroll
    for (int k = 0; k < 7; ++k) env += ea[e * 7 + k] * We[k * 64 + c];
    env = fmaxf(env, 0.0f);
    int uu = u[e], vv = v[e];
    float msg = hsrc[uu * 64 + c] + env;
    if (keep_in) msg *= keep_in[vv] * keep_out[uu];
    if (msg != 0.0f) atomicAdd(&agg[vv * 64 + c], msg);
}

__global__ void agg16_kernel(const float* __restrict__ asrc, const float* __restrict__ ea,
                             const float* __restrict__ We, const float* __restrict__ be,
                             const int* __restrict__ u, const int* __restrict__ v,
                             float* __restrict__ agg, int E) {
    int gid = blockIdx.x * blockDim.x + threadIdx.x;
    if (gid >= E * 16) return;
    int e = gid >> 4, c = gid & 15;
    float env = be[c];
#pragma unroll
    for (int k = 0; k < 7; ++k) env += ea[e * 7 + k] * We[k * 16 + c];
    env = fmaxf(env, 0.0f);
    int uu = u[e], vv = v[e];
    float msg = asrc[uu * 16 + c] + env;
    if (msg != 0.0f) atomicAdd(&agg[vv * 16 + c], msg);
}

// ---------------------------------------------------------------- logits (K=32 -> 2) + gumbel-hard keep flag
__global__ void keep_kernel(const float* __restrict__ a, const float* __restrict__ agg,
                            const float* __restrict__ W2, const float* __restrict__ b2,
                            float* __restrict__ keep, int n, unsigned seed) {
    int i = blockIdx.x * blockDim.x + threadIdx.x;
    if (i >= n) return;
    float l0 = b2[0], l1 = b2[1];
#pragma unroll
    for (int k = 0; k < 16; ++k) {
        float av = a[i * 16 + k];
        l0 += av * W2[k * 2 + 0];
        l1 += av * W2[k * 2 + 1];
    }
#pragma unroll
    for (int k = 0; k < 16; ++k) {
        float gv = agg[i * 16 + k];
        l0 += gv * W2[(16 + k) * 2 + 0];
        l1 += gv * W2[(16 + k) * 2 + 1];
    }
    float g0 = gumbelf(splitmix64(((unsigned long long)seed << 32) | (unsigned)(i * 2 + 0)));
    float g1 = gumbelf(splitmix64(((unsigned long long)seed << 32) | (unsigned)(i * 2 + 1)));
    keep[i] = (l0 + g0 >= l1 + g1) ? 1.0f : 0.0f;   // forward of straight-through hard sample
}

// ---------------------------------------------------------------- shared fragment helpers
__device__ __forceinline__ v8f wmma_tile(const _Float16 (*As)[40], const _Float16 (*Bs)[40],
                                         int waveRow, int waveCol, int lm, int lh, v8f acc) {
    v16h a, b;
    const int ar = (waveRow << 4) + lm;
    const int kb = lh << 3;        // A: lanes 0-15: K 0..7/16..23 ; lanes 16-31: K 8..15/24..31
#pragma unroll
    for (int i = 0; i < 8; ++i) {
        a[i]     = As[ar][kb + i];
        a[i + 8] = As[ar][16 + kb + i];
    }
    const int bn  = (waveCol << 4) + lm;
    const int kbb = lh << 3;       // B: lanes 0-15: K 0..7/16..23 ; lanes 16-31: K 8..15/24..31
#pragma unroll
    for (int i = 0; i < 8; ++i) {
        b[i]     = Bs[bn][kbb + i];
        b[i + 8] = Bs[bn][16 + kbb + i];
    }
    return __builtin_amdgcn_wmma_f32_16x16x32_f16(false, a, false, b, (short)0, acc,
                                                  false, false);
}

// ---------------------------------------------------------------- templated WMMA GEMM (f32 A, K multiples of 32)
// out[nrows, CT*16] = act( concat(A0[:,:K0], A1[:,:K1]) @ W + bias ) (+ residual)
// block = 128 threads (4 wave32); block tile = RT*16 rows x CT*16 cols (RT*CT == 4).
template <int RT, int CT, int K0, int K1, bool RELU, bool RES>
__global__ void __launch_bounds__(128)
gemm_t(const float* __restrict__ A0, const float* __restrict__ A1,
       const _Float16* __restrict__ W, const float* __restrict__ bias,
       const float* __restrict__ res, float* __restrict__ out, int nrows) {
    constexpr int NCOLS = CT * 16;
    constexpr int KTOT  = K0 + K1;
    constexpr int NKC   = KTOT / 32;
    static_assert(K0 % 32 == 0 && K1 % 32 == 0, "clean K only");

    __shared__ __align__(16) _Float16 As[RT * 16][40];   // row-major, 80B row stride
    __shared__ __align__(16) _Float16 Bs[NCOLS][40];     // column-major: Bs[n][k]

    const int tid  = threadIdx.x;
    const int wave = tid >> 5, lane = tid & 31;
    const int lm = lane & 15, lh = lane >> 4;
    const int waveRow = wave % RT, waveCol = wave / RT;
    const int rowBase = blockIdx.x * (RT * 16);
    const bool fullTile = (rowBase + RT * 16) <= nrows;

    v8f acc = {};
#pragma unroll
    for (int kcI = 0; kcI < NKC; ++kcI) {
        const int kc = kcI * 32;
        // ---- stage A tile (f32 -> f16); chunk lies fully in one segment
        const float* src = (kc < K0) ? A0 : A1;
        const int    sK  = (kc < K0) ? K0 : K1;
        const int    kco = (kc < K0) ? kc : (kc - K0);
        if (fullTile) {
#pragma unroll
            for (int it = 0; it < (RT * 16 * 32) / (128 * 4); ++it) {
                int g = tid + it * 128;          // group of 4 consecutive floats
                int r = g >> 3;                  // 8 groups per 32-wide row
                int k = (g & 7) << 2;
                const float4 f = *(const float4*)(src + (size_t)(rowBase + r) * sK + kco + k);
                v4h hv;
                hv[0] = (_Float16)f.x; hv[1] = (_Float16)f.y;
                hv[2] = (_Float16)f.z; hv[3] = (_Float16)f.w;
                *(v4h*)&As[r][k] = hv;
            }
        } else {
            for (int idx = tid; idx < RT * 16 * 32; idx += 128) {
                int r = idx >> 5, k = idx & 31;
                int gr = rowBase + r;
                As[r][k] = (gr < nrows) ? (_Float16)src[(size_t)gr * sK + kco + k]
                                        : (_Float16)0.0f;
            }
        }
        // ---- stage B tile, transposed into Bs[n][k]
#pragma unroll
        for (int it = 0; it < (NCOLS * 32) / (128 * 4); ++it) {
            int g = tid + it * 128;              // group: 4 consecutive k for one n
            int n = g >> 3;
            int k = (g & 7) << 2;
            v4h hv;
#pragma unroll
            for (int j = 0; j < 4; ++j) hv[j] = W[(kc + k + j) * NCOLS + n];
            *(v4h*)&Bs[n][k] = hv;
        }
        __syncthreads();
        acc = wmma_tile(As, Bs, waveRow, waveCol, lm, lh, acc);
        __syncthreads();
    }
    // ---- epilogue.  C/D: lane<16 -> M=r, N=lane ; lane>=16 -> M=8+r, N=lane-16
    const int n  = (waveCol << 4) + lm;
    const int mb = (waveRow << 4) + (lh << 3);
    if (fullTile) {
#pragma unroll
        for (int r = 0; r < 8; ++r) {
            int gr = rowBase + mb + r;
            float v0 = acc[r] + bias[n];
            if (RELU) v0 = fmaxf(v0, 0.0f);
            if (RES)  v0 += res[(size_t)gr * NCOLS + n];
            out[(size_t)gr * NCOLS + n] = v0;
        }
    } else {
#pragma unroll
        for (int r = 0; r < 8; ++r) {
            int gr = rowBase + mb + r;
            if (gr < nrows) {
                float v0 = acc[r] + bias[n];
                if (RELU) v0 = fmaxf(v0, 0.0f);
                if (RES)  v0 += res[(size_t)gr * NCOLS + n];
                out[(size_t)gr * NCOLS + n] = v0;
            }
        }
    }
}

// ---------------------------------------------------------------- f16-A WMMA GEMM (pre-padded A [nrows, KA])
// out[nrows, 64] = relu(A @ W + bias); block tile 16 rows x 64 cols.
template <int KA>
__global__ void __launch_bounds__(128)
gemm_h16(const _Float16* __restrict__ A, const _Float16* __restrict__ W,
         const float* __restrict__ bias, float* __restrict__ out, int nrows) {
    constexpr int NCOLS = 64;
    constexpr int NKC = KA / 32;
    static_assert(KA % 32 == 0, "KA must be a multiple of 32");

    __shared__ __align__(16) _Float16 As[16][40];
    __shared__ __align__(16) _Float16 Bs[NCOLS][40];

    const int tid  = threadIdx.x;
    const int wave = tid >> 5, lane = tid & 31;
    const int lm = lane & 15, lh = lane >> 4;
    const int rowBase = blockIdx.x * 16;
    const bool fullTile = (rowBase + 16) <= nrows;

    v8f acc = {};
#pragma unroll
    for (int kcI = 0; kcI < NKC; ++kcI) {
        const int kc = kcI * 32;
        // ---- stage A (already f16, already padded): one v4h copy per thread
        {
            int g = tid;                         // 128 groups of 4 halves = 16x32 tile
            int r = g >> 3;
            int k = (g & 7) << 2;
            if (fullTile) {
                *(v4h*)&As[r][k] = *(const v4h*)(A + (size_t)(rowBase + r) * KA + kc + k);
            } else {
                int gr = rowBase + r;
                v4h hv;
#pragma unroll
                for (int j = 0; j < 4; ++j)
                    hv[j] = (gr < nrows) ? A[(size_t)gr * KA + kc + k + j] : (_Float16)0.0f;
                *(v4h*)&As[r][k] = hv;
            }
        }
        // ---- stage B transposed into Bs[n][k]
#pragma unroll
        for (int it = 0; it < (NCOLS * 32) / (128 * 4); ++it) {
            int g = tid + it * 128;
            int n = g >> 3;
            int k = (g & 7) << 2;
            v4h hv;
#pragma unroll
            for (int j = 0; j < 4; ++j) hv[j] = W[(kc + k + j) * NCOLS + n];
            *(v4h*)&Bs[n][k] = hv;
        }
        __syncthreads();
        acc = wmma_tile(As, Bs, 0, wave, lm, lh, acc);
        __syncthreads();
    }
    const int n  = (wave << 4) + lm;
    const int mb = lh << 3;
#pragma unroll
    for (int r = 0; r < 8; ++r) {
        int gr = rowBase + mb + r;
        if (fullTile || gr < nrows) {
            float v0 = fmaxf(acc[r] + bias[n], 0.0f);
            out[(size_t)gr * NCOLS + n] = v0;
        }
    }
}

// ---------------------------------------------------------------- segment softmax + attention pooling
__global__ void segmax_kernel(const float* __restrict__ gate, const int* __restrict__ batch,
                              unsigned* __restrict__ gmaxk, int n) {
    int gid = blockIdx.x * blockDim.x + threadIdx.x;
    if (gid >= n * 64) return;
    int i = gid >> 6, c = gid & 63;
    atomicMax(&gmaxk[batch[i] * 64 + c], enc_f(gate[gid]));
}
__global__ void segsum_kernel(const float* __restrict__ gate, const int* __restrict__ batch,
                              const unsigned* __restrict__ gmaxk, float* __restrict__ gsum, int n) {
    int gid = blockIdx.x * blockDim.x + threadIdx.x;
    if (gid >= n * 64) return;
    int i = gid >> 6, c = gid & 63;
    int g = batch[i];
    float e = __expf(gate[gid] - dec_f(gmaxk[g * 64 + c]));
    atomicAdd(&gsum[g * 64 + c], e);
}
__global__ void pool_kernel(const float* __restrict__ gate, const float* __restrict__ emb,
                            const int* __restrict__ batch, const unsigned* __restrict__ gmaxk,
                            const float* __restrict__ gsum, float* __restrict__ pooled, int n) {
    int gid = blockIdx.x * blockDim.x + threadIdx.x;
    if (gid >= n * 64) return;
    int i = gid >> 6, c = gid & 63;
    int g = batch[i];
    float e = __expf(gate[gid] - dec_f(gmaxk[g * 64 + c]));
    float score = e / gsum[g * 64 + c];
    atomicAdd(&pooled[g * 64 + c], score * emb[gid]);
}

// ---------------------------------------------------------------- launch
extern "C" void kernel_launch(void* const* d_in, const int* in_sizes, int n_in,
                              void* d_out, int out_size, void* d_ws, size_t ws_size,
                              hipStream_t stream) {
    (void)n_in; (void)ws_size;
    const float* x      = (const float*)d_in[0];
    const float* eattr  = (const float*)d_in[1];
    const float* Wn     = (const float*)d_in[2];  const float* bn  = (const float*)d_in[3];
    const float* We_env = (const float*)d_in[4];  const float* be_env = (const float*)d_in[5];
    const float* We_act = (const float*)d_in[6];  const float* be_act = (const float*)d_in[7];
    const float* ln_g   = (const float*)d_in[8];  const float* ln_b = (const float*)d_in[9];
    const float* Wa1 = (const float*)d_in[10]; const float* ba1 = (const float*)d_in[11];
    const float* Wa2 = (const float*)d_in[12]; const float* ba2 = (const float*)d_in[13];
    const float* Wo1 = (const float*)d_in[14]; const float* bo1 = (const float*)d_in[15];
    const float* Wo2 = (const float*)d_in[16]; const float* bo2 = (const float*)d_in[17];
    const float* Wc[3]  = {(const float*)d_in[18], (const float*)d_in[20], (const float*)d_in[22]};
    const float* bc[3]  = {(const float*)d_in[19], (const float*)d_in[21], (const float*)d_in[23]};
    const float* Wd1 = (const float*)d_in[24]; const float* bd1 = (const float*)d_in[25];
    const float* Wd2 = (const float*)d_in[26]; const float* bd2 = (const float*)d_in[27];
    const float* Wg1 = (const float*)d_in[28]; const float* bg1 = (const float*)d_in[29];
    const float* Wg2 = (const float*)d_in[30]; const float* bg2 = (const float*)d_in[31];
    const int* eidx  = (const int*)d_in[32];
    const int* batch = (const int*)d_in[33];

    const int N = in_sizes[33];
    const int E = in_sizes[32] / 2;
    const int G = (out_size - N * 64) / 64;
    const int* u = eidx;            // edge_index[0]
    const int* v = eidx + E;        // edge_index[1]

    float* emb    = (float*)d_out;                   // graph_emb [N,64]
    float* pooled = (float*)d_out + (size_t)N * 64;  // pooled [G,64]

    // ---- carve workspace
    char* wsb = (char*)d_ws; size_t off = 0;
    auto take = [&](size_t bytes) -> void* {
        void* p = wsb + off; off += (bytes + 255) & ~(size_t)255; return p;
    };
    float* h      = (float*)take((size_t)N * 64 * 4);
    float* hn     = (float*)take((size_t)N * 64 * 4);
    float* agg64b = (float*)take((size_t)N * 64 * 4);
    float* a_in   = (float*)take((size_t)N * 16 * 4);
    float* a_out  = (float*)take((size_t)N * 16 * 4);
    float* agg16b = (float*)take((size_t)N * 16 * 4);
    float* keep_in  = (float*)take((size_t)N * 4);
    float* keep_out = (float*)take((size_t)N * 4);
    unsigned* gmaxk = (unsigned*)take((size_t)G * 64 * 4);
    float* gsum     = (float*)take((size_t)G * 64 * 4);
    _Float16* xh    = (_Float16*)take((size_t)N * 160 * 2);   // x padded K 153->160
    _Float16* Wn_h  = (_Float16*)take(160 * 64 * 2);          // Wn padded to 160 rows
    _Float16* Wa1_h = (_Float16*)take(128 * 16 * 2);
    _Float16* Wo1_h = (_Float16*)take(128 * 16 * 2);
    _Float16* Wc_h[3];
    for (int i = 0; i < 3; ++i) Wc_h[i] = (_Float16*)take(128 * 64 * 2);
    _Float16* Wd1_h = (_Float16*)take(64 * 64 * 2);
    _Float16* Wd2_h = (_Float16*)take(64 * 64 * 2);
    _Float16* Wg1_h = (_Float16*)take(64 * 64 * 2);
    _Float16* Wg2_h = (_Float16*)take(64 * 64 * 2);

    auto cdiv = [](int a, int b) { return (a + b - 1) / b; };
    auto cvt = [&](const float* src, _Float16* dst, int n) {
        cvt_f16<<<cdiv(n, 256), 256, 0, stream>>>(src, dst, n);
    };
    auto zero = [&](float* p, int n) {
        fill_f32<<<cdiv(n, 256), 256, 0, stream>>>(p, 0.0f, n);
    };

    // ---- weight conversion f32 -> f16 (tiny) + zero-padded inputs
    cvt(Wn, Wn_h, 153 * 64);
    fill_f16<<<cdiv(7 * 64, 256), 256, 0, stream>>>(Wn_h + 153 * 64, 7 * 64);  // pad rows
    cvt(Wa1, Wa1_h, 128 * 16);
    cvt(Wo1, Wo1_h, 128 * 16);
    for (int i = 0; i < 3; ++i) cvt(Wc[i], Wc_h[i], 128 * 64);
    cvt(Wd1, Wd1_h, 64 * 64);
    cvt(Wd2, Wd2_h, 64 * 64);
    cvt(Wg1, Wg1_h, 64 * 64);
    cvt(Wg2, Wg2_h, 64 * 64);
    padcvt_f16<<<cdiv(N * 160, 256), 256, 0, stream>>>(x, xh, N, 153, 160);

    const int g16 = cdiv(N, 16);    // RT=1 tiles
    const int g64 = cdiv(N, 64);    // RT=4 tiles

    // ---- h = relu(x @ Wn + bn)  via padded f16 A [N,160]
    gemm_h16<160><<<g16, 128, 0, stream>>>(xh, Wn_h, bn, h, N);

    const int aggBlocks64 = cdiv(E * 64, 256);
    const int aggBlocks16 = cdiv(E * 16, 256);
    const int lnBlocks = cdiv(N, 8);

    for (int layer = 0; layer < 3; ++layer) {
        // hn = LN(h)
        ln_kernel<<<lnBlocks, 256, 0, stream>>>(h, ln_g, ln_b, hn, N);
        // agg1 = segsum(hn[u] + env_ea)  (shared by both action branches)
        zero(agg64b, N * 64);
        agg64_kernel<<<aggBlocks64, 256, 0, stream>>>(hn, eattr, We_env, be_env, u, v,
                                                      nullptr, nullptr, agg64b, E);
        // a_in = relu([hn|agg1] @ Wa1 + ba1)
        gemm_t<4, 1, 64, 64, true, false><<<g64, 128, 0, stream>>>(hn, agg64b, Wa1_h, ba1,
                                                                   nullptr, a_in, N);
        // agg2 = segsum(a_in[u] + act_ea); in_logits -> keep_in
        zero(agg16b, N * 16);
        agg16_kernel<<<aggBlocks16, 256, 0, stream>>>(a_in, eattr, We_act, be_act, u, v, agg16b, E);
        keep_kernel<<<cdiv(N, 256), 256, 0, stream>>>(a_in, agg16b, Wa2, ba2, keep_in, N,
                                                      0x1234u + 2u * layer);
        // a_out branch
        gemm_t<4, 1, 64, 64, true, false><<<g64, 128, 0, stream>>>(hn, agg64b, Wo1_h, bo1,
                                                                   nullptr, a_out, N);
        zero(agg16b, N * 16);
        agg16_kernel<<<aggBlocks16, 256, 0, stream>>>(a_out, eattr, We_act, be_act, u, v, agg16b, E);
        keep_kernel<<<cdiv(N, 256), 256, 0, stream>>>(a_out, agg16b, Wo2, bo2, keep_out, N,
                                                      0x1235u + 2u * layer);
        // gated aggregation: agg4 = segsum(keep_in[v]*keep_out[u] * (hn[u] + env_ea))
        zero(agg64b, N * 64);
        agg64_kernel<<<aggBlocks64, 256, 0, stream>>>(hn, eattr, We_env, be_env, u, v,
                                                      keep_in, keep_out, agg64b, E);
        // h = h + relu([hn|agg4] @ Wc + bc)
        gemm_t<1, 4, 64, 64, true, true><<<g16, 128, 0, stream>>>(hn, agg64b, Wc_h[layer],
                                                                  bc[layer], h, h, N);
    }

    // ---- decoder + gating head
    ln_kernel<<<lnBlocks, 256, 0, stream>>>(h, ln_g, ln_b, hn, N);
    gemm_t<1, 4, 64, 0, true,  false><<<g16, 128, 0, stream>>>(hn, nullptr, Wd1_h, bd1,
                                                               nullptr, agg64b, N);  // t
    gemm_t<1, 4, 64, 0, false, false><<<g16, 128, 0, stream>>>(agg64b, nullptr, Wd2_h, bd2,
                                                               nullptr, emb, N);     // graph_emb
    gemm_t<1, 4, 64, 0, true,  false><<<g16, 128, 0, stream>>>(emb, nullptr, Wg1_h, bg1,
                                                               nullptr, hn, N);      // g1
    gemm_t<1, 4, 64, 0, false, false><<<g16, 128, 0, stream>>>(hn, nullptr, Wg2_h, bg2,
                                                               nullptr, agg64b, N);  // gate

    // ---- segment softmax over batch, attention pooling
    fill_u32<<<cdiv(G * 64, 256), 256, 0, stream>>>(gmaxk, 0u, G * 64);  // key(-inf) ~ 0
    zero(gsum, G * 64);
    zero(pooled, G * 64);
    int segBlocks = cdiv(N * 64, 256);
    segmax_kernel<<<segBlocks, 256, 0, stream>>>(agg64b, batch, gmaxk, N);
    segsum_kernel<<<segBlocks, 256, 0, stream>>>(agg64b, batch, gmaxk, gsum, N);
    pool_kernel<<<segBlocks, 256, 0, stream>>>(agg64b, emb, batch, gmaxk, gsum, pooled, N);
}
